// SSGSetAbstraction_46145128628929
// MI455X (gfx1250) — matrix operand order repaired and checked by reading, not compiled, not run on hardware
//
#include <hip/hip_runtime.h>

// ---------------- problem constants ----------------
constexpr int NPTS  = 4096;
constexpr int MCENT = 1024;            // N/4
constexpr int KG    = 64;              // group size
constexpr int CINC  = 64;
constexpr int COUTC = 128;
constexpr int BATCH = 16;
constexpr int NTOT  = BATCH * MCENT * KG;   // 1<<20 columns
constexpr int GBLK  = NTOT / 128;           // 8192 gemm blocks (128 cols/block)
constexpr float RAD2 = 0.2f * 0.2f;

// ---------------- vector types ----------------
typedef __attribute__((ext_vector_type(2)))  float        v2f;
typedef __attribute__((ext_vector_type(8)))  float        v8f;
typedef __attribute__((ext_vector_type(4)))  int          v4i;
typedef __attribute__((ext_vector_type(8)))  unsigned int v8u;
typedef __attribute__((ext_vector_type(16))) __bf16       v16bf;

static __device__ inline v8f vzero() {
  v8f z = {0.f,0.f,0.f,0.f,0.f,0.f,0.f,0.f};
  return z;
}

// bf16 helpers (round-to-nearest-even pack)
static __device__ inline unsigned int f2bf(float f) {
  unsigned int u = __builtin_bit_cast(unsigned int, f);
  unsigned int r = u + 0x7FFFu + ((u >> 16) & 1u);
  return r >> 16;
}
static __device__ inline unsigned int pack2(float lo, float hi) {
  return f2bf(lo) | (f2bf(hi) << 16);
}
static __device__ inline float bf2f(unsigned int h) {
  return __builtin_bit_cast(float, h << 16);
}

// ---------------- gfx1250 async global->LDS copy ----------------
// GLOBAL_LOAD_ASYNC_TO_LDS_B128: ASYNCcnt-tracked DMA into LDS, no VGPR
// round trip. Builtin signature (from hipcc diagnostic): typed v4i
// pointers in AS1 (global) / AS3 (LDS). Guarded so a toolchain without
// the builtin still compiles (falls back to a plain staged copy).
#if defined(__has_builtin)
#if __has_builtin(__builtin_amdgcn_global_load_async_to_lds_b128) && \
    __has_builtin(__builtin_amdgcn_s_wait_asynccnt)
#define HAVE_ASYNC_LDS 1
#endif
#endif

#ifdef HAVE_ASYNC_LDS
typedef __attribute__((address_space(1))) v4i gv4i;
typedef __attribute__((address_space(3))) v4i sv4i;
#endif

__device__ inline void async_copy_f32(float* lds_dst, const float* gsrc,
                                      int count, int t, int nthreads) {
#ifdef HAVE_ASYNC_LDS
  for (int i = t * 4; i < count; i += nthreads * 4) {
    // generic VA == global VA; low 32 bits of generic VA == LDS byte offset
    gv4i* gp = (gv4i*)(unsigned long long)(size_t)(gsrc + i);
    sv4i* lp = (sv4i*)(unsigned int)(size_t)(lds_dst + i);
    __builtin_amdgcn_global_load_async_to_lds_b128(gp, lp, 0, 0);
  }
#else
  for (int i = t; i < count; i += nthreads) lds_dst[i] = gsrc[i];
#endif
}

__device__ inline void async_copy_wait() {
#ifdef HAVE_ASYNC_LDS
  __builtin_amdgcn_s_wait_asynccnt(0);
#endif
}

// ------------------------------------------------------------------
// Kernel 1: farthest point sampling. One 256-thread block per batch.
// xyz cached in LDS (48KB, async DMA); per-thread min-dist in 16 regs.
// Matches jax scan semantics: first emitted index is 0, argmax ties
// resolved to smallest index.
// ------------------------------------------------------------------
__global__ void fps_kernel(const float* __restrict__ pts,
                           float* __restrict__ cent,
                           float* __restrict__ out_cent) {
  __shared__ __align__(16) float xs[NPTS];
  __shared__ __align__(16) float ys[NPTS];
  __shared__ __align__(16) float zs[NPTS];
  __shared__ float redv[256];
  __shared__ int   redi[256];
  __shared__ int   s_li;
  const int b = blockIdx.x;
  const int t = threadIdx.x;
  const float* px = pts + (size_t)b * 3 * NPTS;
  async_copy_f32(xs, px, NPTS, t, 256);
  async_copy_f32(ys, px + NPTS, NPTS, t, 256);
  async_copy_f32(zs, px + 2 * NPTS, NPTS, t, 256);
  float dist[16];
#pragma unroll
  for (int j = 0; j < 16; j++) dist[j] = 1e10f;
  if (t == 0) s_li = 0;
  async_copy_wait();
  __syncthreads();

  for (int m = 0; m < MCENT; m++) {
    const int li = s_li;
    if (t == 0) {
      size_t o = ((size_t)b * MCENT + m) * 3;
      float cx = xs[li], cy = ys[li], cz = zs[li];
      cent[o] = cx; cent[o + 1] = cy; cent[o + 2] = cz;
      out_cent[o] = cx; out_cent[o + 1] = cy; out_cent[o + 2] = cz;
    }
    const float lx = xs[li], ly = ys[li], lz = zs[li];
    float bv = -1.f; int bi = 0;
#pragma unroll
    for (int j = 0; j < 16; j++) {
      int i = j * 256 + t;
      float dx = xs[i] - lx, dy = ys[i] - ly, dz = zs[i] - lz;
      float d = dx * dx + dy * dy + dz * dz;
      d = fminf(dist[j], d);
      dist[j] = d;
      if (d > bv) { bv = d; bi = i; }
    }
    redv[t] = bv; redi[t] = bi;
    __syncthreads();
    for (int s = 128; s > 0; s >>= 1) {
      if (t < s) {
        float ov = redv[t + s]; int oi = redi[t + s];
        if (ov > redv[t] || (ov == redv[t] && oi < redi[t])) {
          redv[t] = ov; redi[t] = oi;
        }
      }
      __syncthreads();
    }
    if (t == 0) s_li = redi[0];
    __syncthreads();
  }
}

// ------------------------------------------------------------------
// Kernel 2: ball query. One wave32 per centroid (8 waves/block),
// xyz in LDS (async DMA). Ballot + prefix-popc -> first K in index
// order, padded with the first hit. Early exit once cnt>=K.
// ------------------------------------------------------------------
__global__ void ballq_kernel(const float* __restrict__ pts,
                             const float* __restrict__ cent,
                             int* __restrict__ gidx) {
  __shared__ __align__(16) float xs[NPTS];
  __shared__ __align__(16) float ys[NPTS];
  __shared__ __align__(16) float zs[NPTS];
  __shared__ int gbuf[8][KG];
  const int t = threadIdx.x;
  const int w = t >> 5;
  const int lane = t & 31;
  const int blk = blockIdx.x;
  const int b = blk >> 7;             // MCENT/8 = 128 groups per batch
  const int m0 = (blk & 127) * 8;
  const float* px = pts + (size_t)b * 3 * NPTS;
  async_copy_f32(xs, px, NPTS, t, 256);
  async_copy_f32(ys, px + NPTS, NPTS, t, 256);
  async_copy_f32(zs, px + 2 * NPTS, NPTS, t, 256);
  async_copy_wait();
  __syncthreads();

  const int m = m0 + w;
  const size_t co = ((size_t)b * MCENT + m) * 3;
  const float cx = cent[co], cy = cent[co + 1], cz = cent[co + 2];
  const unsigned int ltmask = (1u << lane) - 1u;
  int cnt = 0;
  for (int base = 0; base < NPTS && cnt < KG; base += 32) {
    const int i = base + lane;
    float dx = xs[i] - cx, dy = ys[i] - cy, dz = zs[i] - cz;
    bool within = (dx * dx + dy * dy + dz * dz) <= RAD2;
    unsigned int msk = (unsigned int)__ballot(within);
    int pos = cnt + __popc(msk & ltmask);
    if (within && pos < KG) gbuf[w][pos] = i;
    cnt += __popc(msk);
  }
  __syncthreads();
  int c2 = cnt < KG ? cnt : KG;
  int first = (c2 > 0) ? gbuf[w][0] : 0;
  for (int j = c2 + lane; j < KG; j += 32) gbuf[w][j] = first;
  __syncthreads();
  int* gout = gidx + ((size_t)b * MCENT + m0) * KG;
  for (int i = t; i < 8 * KG; i += 256) gout[i] = gbuf[i >> 6][i & 63];
}

// ------------------------------------------------------------------
// shared GEMM epilogue: bias, bf16 store ([n][64] point-major),
// deterministic per-channel partial sum / sumsq (no atomics).
// C/D layout: VGPR v -> rows v (lanes 0-15) and v+8 (lanes 16-31).
// ------------------------------------------------------------------
__device__ inline void tile_epilogue(const v8f& acc, int r, size_t n,
                                     int half, int l16, int w,
                                     const float* __restrict__ bias,
                                     unsigned short* __restrict__ y,
                                     float (*ssum)[64], float (*ssq)[64]) {
#pragma unroll
  for (int v = 0; v < 8; v++) {
    int o = 16 * r + v + 8 * half;
    float val = acc[v] + bias[o];
    y[n * 64 + o] = (unsigned short)f2bf(val);
    float s = val, q = val * val;
#pragma unroll
    for (int mk = 1; mk < 16; mk <<= 1) {
      s += __shfl_xor(s, mk, 32);
      q += __shfl_xor(q, mk, 32);
    }
    if (l16 == 0) { ssum[w][o] = s; ssq[w][o] = q; }
  }
}

// 16-bit A fragment from LDS-staged bf16-pair-packed weights
// (row o, 64 K-columns, K-chunk kk of 32).
// A 16x32 layout: VGPR j<4 -> K = 2j+8*half (+1); j>=4 -> K = 16+2(j-4)+8*half.
__device__ inline v16bf make_w_frag_lds(const unsigned int* wp, int o,
                                        int kk, int half) {
  unsigned int ak[8];
#pragma unroll
  for (int j = 0; j < 8; j++) {
    int kl = (j < 4) ? (2 * j + 8 * half) : (16 + 2 * (j - 4) + 8 * half);
    int c = 32 * kk + kl;              // always even
    ak[j] = wp[(o * 64 + c) >> 1];
  }
  v8u av = {ak[0], ak[1], ak[2], ak[3], ak[4], ak[5], ak[6], ak[7]};
  return __builtin_bit_cast(v16bf, av);
}

// 16-bit B fragment (32x16) from bf16 activations with fused BN+ReLU.
// B layout: lanes 0-15 = columns, VGPR v<4 packs K = 8*half + 2v, +1;
// v>=4 packs K = 16 + 8*half + 2(v-4), +1.
__device__ inline v16bf make_act_frag(const unsigned short* __restrict__ y,
                                      size_t n, int kk, int half,
                                      const float* lA, const float* lB) {
  const int c0 = 32 * kk + 8 * half;
  const uint4* pa = reinterpret_cast<const uint4*>(y + n * 64 + c0);
  uint4 q0 = pa[0], q1 = pa[2];
  unsigned int wd[8] = {q0.x, q0.y, q0.z, q0.w, q1.x, q1.y, q1.z, q1.w};
  unsigned int pk[8];
#pragma unroll
  for (int e = 0; e < 8; e++) {
    int c = c0 + 2 * e + ((e >= 4) ? 8 : 0);
    float v0 = fmaxf(bf2f(wd[e] & 0xffffu) * lA[c] + lB[c], 0.f);
    float v1 = fmaxf(bf2f(wd[e] >> 16) * lA[c + 1] + lB[c + 1], 0.f);
    pk[e] = pack2(v0, v1);
  }
  v8u pv = {pk[0], pk[1], pk[2], pk[3], pk[4], pk[5], pk[6], pk[7]};
  return __builtin_bit_cast(v16bf, pv);
}

// ------------------------------------------------------------------
// Kernel 3: layer-1 GEMM, fused gather. fp32 WMMA 16x16x4.
// Column n = (b*MCENT+m)*KG + k. Channels: 0..2 rel-xyz, 3..66 feats,
// 67 zero pad (17 K-steps of 4). w1 staged in LDS padded to 68 cols
// so per-step A loads are 8B-aligned ds_load_b64 (col 67 == 0 pad).
// ------------------------------------------------------------------
__device__ inline float chan1(int c, float r0, float r1, float r2,
                              const float* __restrict__ fb) {
  if (c < 3) return (c == 0) ? r0 : ((c == 1) ? r1 : r2);
  if (c < 67) return fb[(size_t)(c - 3) * NPTS];
  return 0.f;
}

__global__ void gemm1_kernel(const float* __restrict__ pts,
                             const float* __restrict__ pf,
                             const int* __restrict__ gidx,
                             const float* __restrict__ cent,
                             const float* __restrict__ w1,
                             const float* __restrict__ b1,
                             unsigned short* __restrict__ y1,
                             float* __restrict__ partials) {
  __shared__ float ssum[8][64];
  __shared__ float ssq[8][64];
  __shared__ __align__(8) float lw[64 * 68];
  const int t = threadIdx.x;
  const int w = t >> 5;
  const int lane = t & 31;
  const int half = lane >> 4;
  const int l16 = lane & 15;
  for (int i = t; i < 64 * 68; i += 256) {
    int o = i / 68, c = i - o * 68;
    lw[i] = (c < 67) ? w1[o * 67 + c] : 0.f;
  }
  const size_t n = (size_t)blockIdx.x * 128 + w * 16 + l16;
  const int b = (int)(n >> 16);
  const int mm = (int)(n >> 6) & (MCENT - 1);
  const int p = gidx[n];
  const float* px = pts + (size_t)b * 3 * NPTS;
  const size_t co = ((size_t)b * MCENT + mm) * 3;
  const float r0 = px[p] - cent[co];
  const float r1 = px[NPTS + p] - cent[co + 1];
  const float r2 = px[2 * NPTS + p] - cent[co + 2];
  const float* fb = pf + (size_t)b * CINC * NPTS + p;
  __syncthreads();

  v8f a0 = vzero(), a1 = vzero(), a2 = vzero(), a3 = vzero();
  for (int kk = 0; kk < 17; kk++) {
    const int c0 = 4 * kk + 2 * half;
    float bx = chan1(c0, r0, r1, r2, fb);
    float by = chan1(c0 + 1, r0, r1, r2, fb);
    v2f bf = {bx, by};
#define L1_STEP(R, ACC)                                                        \
    {                                                                          \
      int o = 16 * R + l16;                                                    \
      v2f af = *(const v2f*)&lw[o * 68 + c0];                                  \
      ACC = __builtin_amdgcn_wmma_f32_16x16x4_f32(false, af, false, bf,        \
                                                  (short)0, ACC, false, false);\
    }
    L1_STEP(0, a0) L1_STEP(1, a1) L1_STEP(2, a2) L1_STEP(3, a3)
#undef L1_STEP
  }
  tile_epilogue(a0, 0, n, half, l16, w, b1, y1, ssum, ssq);
  tile_epilogue(a1, 1, n, half, l16, w, b1, y1, ssum, ssq);
  tile_epilogue(a2, 2, n, half, l16, w, b1, y1, ssum, ssq);
  tile_epilogue(a3, 3, n, half, l16, w, b1, y1, ssum, ssq);
  __syncthreads();
  if (t < 64) {
    float s = 0.f, q = 0.f;
#pragma unroll
    for (int ww = 0; ww < 8; ww++) { s += ssum[ww][t]; q += ssq[ww][t]; }
    partials[(size_t)blockIdx.x * 128 + t] = s;
    partials[(size_t)blockIdx.x * 128 + 64 + t] = q;
  }
}

// ------------------------------------------------------------------
// Kernel 4: fixed-order BN stat reduction -> scale/shift (deterministic).
// bn[0..63] = g*rsqrt(var+eps), bn[64..127] = beta - mean*scale.
// ------------------------------------------------------------------
__global__ void stats_finalize(const float* __restrict__ partials,
                               const float* __restrict__ g,
                               const float* __restrict__ beta,
                               float* __restrict__ bn, int nblocks) {
  __shared__ float rs[256], rq[256];
  const int o = blockIdx.x, t = threadIdx.x;
  float s = 0.f, q = 0.f;
  for (int i = t; i < nblocks; i += 256) {
    s += partials[(size_t)i * 128 + o];
    q += partials[(size_t)i * 128 + 64 + o];
  }
  rs[t] = s; rq[t] = q;
  __syncthreads();
  for (int st = 128; st > 0; st >>= 1) {
    if (t < st) { rs[t] += rs[t + st]; rq[t] += rq[t + st]; }
    __syncthreads();
  }
  if (t == 0) {
    const float inv = 1.f / (float)NTOT;
    float mean = rs[0] * inv;
    float var = rq[0] * inv - mean * mean;
    float sc = g[o] * rsqrtf(var + 1e-5f);
    bn[o] = sc;
    bn[64 + o] = beta[o] - mean * sc;
  }
}

// ------------------------------------------------------------------
// Kernel 5: layer-2 GEMM, bf16 WMMA 16x16x32 with fused BN+ReLU on
// input. Weights pre-packed to bf16 pairs in LDS once per block.
// ------------------------------------------------------------------
__global__ void gemm2_kernel(const unsigned short* __restrict__ y1,
                             const float* __restrict__ w2,
                             const float* __restrict__ b2,
                             const float* __restrict__ bn,
                             unsigned short* __restrict__ y2,
                             float* __restrict__ partials) {
  __shared__ float lA[64], lB[64];
  __shared__ float ssum[8][64];
  __shared__ float ssq[8][64];
  __shared__ unsigned int lw[64 * 32];
  const int t = threadIdx.x;
  const int w = t >> 5;
  const int lane = t & 31;
  const int half = lane >> 4;
  const int l16 = lane & 15;
  if (t < 64) { lA[t] = bn[t]; lB[t] = bn[64 + t]; }
  const float2* w2v = reinterpret_cast<const float2*>(w2);
  for (int i = t; i < 64 * 32; i += 256) {
    float2 f = w2v[i];
    lw[i] = pack2(f.x, f.y);
  }
  __syncthreads();
  const size_t n = (size_t)blockIdx.x * 128 + w * 16 + l16;

  v8f acc[4];
#pragma unroll
  for (int r = 0; r < 4; r++) acc[r] = vzero();
#pragma unroll
  for (int kk = 0; kk < 2; kk++) {
    v16bf bfrag = make_act_frag(y1, n, kk, half, lA, lB);
#pragma unroll
    for (int r = 0; r < 4; r++) {
      v16bf afrag = make_w_frag_lds(lw, 16 * r + l16, kk, half);
      acc[r] = __builtin_amdgcn_wmma_f32_16x16x32_bf16(
          false, afrag, false, bfrag, (short)0, acc[r], false, false);
    }
  }
#pragma unroll
  for (int r = 0; r < 4; r++)
    tile_epilogue(acc[r], r, n, half, l16, w, b2, y2, ssum, ssq);
  __syncthreads();
  if (t < 64) {
    float s = 0.f, q = 0.f;
#pragma unroll
    for (int ww = 0; ww < 8; ww++) { s += ssum[ww][t]; q += ssq[ww][t]; }
    partials[(size_t)blockIdx.x * 128 + t] = s;
    partials[(size_t)blockIdx.x * 128 + 64 + t] = q;
  }
}

// ------------------------------------------------------------------
// Kernel 6: layer-3 GEMM (O=128) + fused max over group dim.
// One wave per group (64 columns = 4 tiles, maxed in registers),
// then 16-lane shuffle-max -> centroid_features[b][o][m].
// ------------------------------------------------------------------
__global__ void gemm3_kernel(const unsigned short* __restrict__ y2,
                             const float* __restrict__ w3,
                             const float* __restrict__ b3,
                             const float* __restrict__ bn,
                             float* __restrict__ out_feat) {
  __shared__ float lA[64], lB[64];
  __shared__ unsigned int lw[128 * 32];
  const int t = threadIdx.x;
  const int w = t >> 5;
  const int lane = t & 31;
  const int half = lane >> 4;
  const int l16 = lane & 15;
  if (t < 64) { lA[t] = bn[t]; lB[t] = bn[64 + t]; }
  const float2* w3v = reinterpret_cast<const float2*>(w3);
  for (int i = t; i < 128 * 32; i += 256) {
    float2 f = w3v[i];
    lw[i] = pack2(f.x, f.y);
  }
  __syncthreads();
  const int grp = blockIdx.x * 8 + w;     // b*MCENT + m
  const int b = grp >> 10;
  const int m = grp & (MCENT - 1);

  v8f vm[8];
#pragma unroll
  for (int ct = 0; ct < 4; ct++) {
    const size_t n = (size_t)grp * 64 + ct * 16 + l16;
    v8f acc[8];
#pragma unroll
    for (int r = 0; r < 8; r++) acc[r] = vzero();
#pragma unroll
    for (int kk = 0; kk < 2; kk++) {
      v16bf bfrag = make_act_frag(y2, n, kk, half, lA, lB);
#pragma unroll
      for (int r = 0; r < 8; r++) {
        v16bf afrag = make_w_frag_lds(lw, 16 * r + l16, kk, half);
        acc[r] = __builtin_amdgcn_wmma_f32_16x16x32_bf16(
            false, afrag, false, bfrag, (short)0, acc[r], false, false);
      }
    }
#pragma unroll
    for (int r = 0; r < 8; r++) {
      if (ct == 0) {
        vm[r] = acc[r];
      } else {
#pragma unroll
        for (int v = 0; v < 8; v++) vm[r][v] = fmaxf(vm[r][v], acc[r][v]);
      }
    }
  }
#pragma unroll
  for (int r = 0; r < 8; r++) {
#pragma unroll
    for (int v = 0; v < 8; v++) {
      float x = vm[r][v];
#pragma unroll
      for (int mk = 1; mk < 16; mk <<= 1) x = fmaxf(x, __shfl_xor(x, mk, 32));
      if (l16 == 0) {
        int o = 16 * r + v + 8 * half;
        out_feat[(size_t)b * COUTC * MCENT + (size_t)o * MCENT + m] = x + b3[o];
      }
    }
  }
}

// ------------------------------------------------------------------
// host-side launcher
// ------------------------------------------------------------------
extern "C" void kernel_launch(void* const* d_in, const int* in_sizes, int n_in,
                              void* d_out, int out_size, void* d_ws,
                              size_t ws_size, hipStream_t stream) {
  (void)in_sizes; (void)n_in; (void)out_size; (void)ws_size;
  const float* pts = (const float*)d_in[0];
  const float* pf  = (const float*)d_in[1];
  const float* w1  = (const float*)d_in[2];
  const float* b1  = (const float*)d_in[3];
  const float* g1  = (const float*)d_in[4];
  const float* be1 = (const float*)d_in[5];
  const float* w2  = (const float*)d_in[6];
  const float* b2  = (const float*)d_in[7];
  const float* g2  = (const float*)d_in[8];
  const float* be2 = (const float*)d_in[9];
  const float* w3  = (const float*)d_in[10];
  const float* b3  = (const float*)d_in[11];

  float* out = (float*)d_out;
  float* out_cent = out;                              // B*M*3 (== view(B,3,M))
  float* out_feat = out + (size_t)BATCH * 3 * MCENT;  // B*COUT*M

  char* ws = (char*)d_ws;
  float* cent = (float*)ws;              ws += (size_t)BATCH * MCENT * 3 * 4;
  int* gidx = (int*)ws;                  ws += (size_t)BATCH * MCENT * KG * 4;
  float* bn1 = (float*)ws;               ws += 128 * 4;
  float* bn2 = (float*)ws;               ws += 128 * 4;
  float* parts = (float*)ws;             ws += (size_t)GBLK * 128 * 4;
  unsigned short* y1 = (unsigned short*)ws; ws += (size_t)NTOT * 64 * 2;
  unsigned short* y2 = (unsigned short*)ws;

  fps_kernel<<<BATCH, 256, 0, stream>>>(pts, cent, out_cent);
  ballq_kernel<<<BATCH * (MCENT / 8), 256, 0, stream>>>(pts, cent, gidx);
  gemm1_kernel<<<GBLK, 256, 0, stream>>>(pts, pf, gidx, cent, w1, b1, y1, parts);
  stats_finalize<<<64, 256, 0, stream>>>(parts, g1, be1, bn1, GBLK);
  gemm2_kernel<<<GBLK, 256, 0, stream>>>(y1, w2, b2, bn1, y2, parts);
  stats_finalize<<<64, 256, 0, stream>>>(parts, g2, be2, bn2, GBLK);
  gemm3_kernel<<<BATCH * MCENT / 8, 256, 0, stream>>>(y2, w3, b3, bn2, out_feat);
}
